// ConditionalCrossAttention_53154515255722
// MI455X (gfx1250) — matrix operand order repaired and checked by reading, not compiled, not run on hardware
//
#include <hip/hip_runtime.h>

#define DIMM 1024
#define KVD  768
#define NH   16
#define HD   64
#define LQ   4096
#define LKV  1024
#define BB   2

typedef __attribute__((ext_vector_type(16))) _Float16 v16h;
typedef __attribute__((ext_vector_type(8)))  float    v8f;
typedef int v4i_g __attribute__((vector_size(16)));   // matches builtin param type

#if defined(__gfx1250__) && __has_builtin(__builtin_amdgcn_global_load_async_to_lds_b128)
#define HAS_ASYNC 1
#else
#define HAS_ASYNC 0
#endif

union FragU { v16h h; unsigned int u[8]; };

// ---- CDNA5 async global->LDS staging (16B per lane), sync fallback ----
__device__ __forceinline__ void stage16(const _Float16* g, _Float16* lds) {
#if HAS_ASYNC
  __builtin_amdgcn_global_load_async_to_lds_b128(
      (__attribute__((address_space(1))) v4i_g*)g,
      (__attribute__((address_space(3))) v4i_g*)lds, 0, 0);
#else
  *(uint4*)lds = *(const uint4*)g;
#endif
}

template <int N>
__device__ __forceinline__ void wait_async() {
#if HAS_ASYNC
#if __has_builtin(__builtin_amdgcn_s_wait_asynccnt)
  __builtin_amdgcn_s_wait_asynccnt(N);
#else
  asm volatile("s_wait_asynccnt %0" ::"i"(N) : "memory");
#endif
#endif
}

// A-fragment (16x32 f16) from LDS, documented CDNA5 layout:
// lane l: M = rowBase + (l&15); half = l>>4; vgprs 0..3: K = 8*half + {0..7},
// vgprs 4..7: K = 16 + 8*half + {0..7}; K-pairs contiguous -> b32 reads.
__device__ __forceinline__ v16h frag_a_lds(const _Float16* tile, int stride,
                                           int rowBase, int kBase, int lane) {
  const int m  = rowBase + (lane & 15);
  const int hs = lane >> 4;
  const _Float16* row = tile + m * stride + kBase;
  FragU f;
#pragma unroll
  for (int j = 0; j < 4; ++j)
    f.u[j] = *(const unsigned int*)(row + 8 * hs + 2 * j);
#pragma unroll
  for (int j = 0; j < 4; ++j)
    f.u[4 + j] = *(const unsigned int*)(row + 16 + 8 * hs + 2 * j);
  return f.h;
}

// B-fragment (32x16 f16): lane holds column N = colBase + (l&15); half = l>>4;
// vgprs 0..3: K = 8*half + {0..7}, vgprs 4..7: K = 16 + 8*half + {0..7}.
// Tile stored [N][K] so reads are contiguous.
__device__ __forceinline__ v16h frag_b_lds(const _Float16* tile, int stride,
                                           int colBase, int kBase, int lane) {
  const int n  = colBase + (lane & 15);
  const int hs = lane >> 4;
  const _Float16* row = tile + n * stride + kBase;
  FragU f;
#pragma unroll
  for (int j = 0; j < 4; ++j)
    f.u[j] = *(const unsigned int*)(row + 8 * hs + 2 * j);
#pragma unroll
  for (int j = 0; j < 4; ++j)
    f.u[4 + j] = *(const unsigned int*)(row + 16 + 8 * hs + 2 * j);
  return f.h;
}

__device__ __forceinline__ v8f wmma_f16(v16h a, v16h b, v8f c) {
  return __builtin_amdgcn_wmma_f32_16x16x32_f16(false, a, false, b,
                                                (short)0, c, false, false);
}

// ---------------- conversion kernels ----------------

__global__ __launch_bounds__(256) void cvt_f32_to_f16_kernel(
    const float* __restrict__ src, _Float16* __restrict__ dst, int n) {
  const int i = blockIdx.x * 256 + threadIdx.x;
  if (i < n) dst[i] = (_Float16)src[i];
}

// W (K x N row-major f32) -> Wt (N x K row-major f16) == W^T
__global__ __launch_bounds__(256) void cvt_transpose_kernel(
    const float* __restrict__ W, _Float16* __restrict__ Wt, int K, int N) {
  const int i = blockIdx.x * 256 + threadIdx.x;
  if (i < K * N) {
    const int k = i / N, n = i - k * N;
    Wt[(size_t)n * K + k] = (_Float16)W[i];
  }
}

// Vf: (B*LKV, DIMM) f32 -> Vh: [B][H][HD][LKV] f16 (pre-transposed per head so
// the attention kernel can async-stage contiguous b128 rows of V^T)
__global__ __launch_bounds__(256) void remap_v_kernel(
    const float* __restrict__ Vf, _Float16* __restrict__ Vh) {
  const int i = blockIdx.x * 256 + threadIdx.x;   // exact: B*LKV*DIMM threads
  const int row = i >> 10, col = i & 1023;
  const int b = row / LKV, pos = row - b * LKV;
  const int head = col >> 6, hd = col & 63;
  Vh[((size_t)(b * NH + head) * HD + hd) * LKV + pos] = (_Float16)Vf[i];
}

// ---------------- RMSNorm + RoPE ----------------
// One block per row. X: (rows, DIMM) f32. out: [B][H][L][HD] f16.
__global__ __launch_bounds__(256) void rmsnorm_rope_kernel(
    const float* __restrict__ X, const float* __restrict__ g,
    const float* __restrict__ cosb, const float* __restrict__ sinb,
    _Float16* __restrict__ out, int L, float scale) {
  __shared__ float xsh[DIMM];
  __shared__ float nsh[DIMM];
  __shared__ float red[256];
  const int row = blockIdx.x;
  const int b = row / L, pos = row - b * L;
  const int tid = threadIdx.x;
  const float* x = X + (size_t)row * DIMM;

  float ss = 0.f;
  for (int j = tid; j < DIMM; j += 256) {
    const float v = x[j];
    xsh[j] = v;
    ss += v * v;
  }
  red[tid] = ss;
  __syncthreads();
  for (int sT = 128; sT > 0; sT >>= 1) {
    if (tid < sT) red[tid] += red[tid + sT];
    __syncthreads();
  }
  const float rstd = rsqrtf(red[0] / (float)DIMM + 1e-6f);
  for (int j = tid; j < DIMM; j += 256) nsh[j] = xsh[j] * rstd * g[j];
  __syncthreads();

  const float* cr = cosb + (size_t)row * HD;
  const float* sr = sinb + (size_t)row * HD;
  for (int j = tid; j < DIMM; j += 256) {
    const int head = j >> 6, hd = j & 63;
    const float rot = (hd < 32) ? -nsh[j + 32] : nsh[j - 32];
    const float v = (nsh[j] * cr[hd] + rot * sr[hd]) * scale;
    out[(((size_t)(b * NH + head)) * L + pos) * HD + hd] = (_Float16)v;
  }
}

// ---------------- WMMA GEMM: C(f32) = A(f16, MxK) * Bt(f16, NxK)^T + bias ----
// Block: 256 threads = 8 waves (2x4), 128x256 tile; wave = 64x64 (16 WMMA per
// K-step vs 8 fragment loads). K staged 32 deep, double-buffered via
// global_load_async_to_lds_b128 + s_wait_asynccnt.
__global__ __launch_bounds__(256) void gemm_f16_wmma_kernel(
    const _Float16* __restrict__ A, const _Float16* __restrict__ Bt,
    const float* __restrict__ bias, float* __restrict__ C, int N, int K) {
  __shared__ _Float16 As[2][128][40];
  __shared__ _Float16 Bs[2][256][40];
  const int tid = threadIdx.x, lane = tid & 31, wave = tid >> 5;
  const int waveM = wave >> 2, waveN = wave & 3;
  const size_t rowBlock = (size_t)blockIdx.y * 128;
  const size_t colBlock = (size_t)blockIdx.x * 256;

  v8f acc[4][4];
#pragma unroll
  for (int im = 0; im < 4; ++im)
#pragma unroll
    for (int jn = 0; jn < 4; ++jn) { v8f z = {}; acc[im][jn] = z; }

  const int nk = K / 32;

  // prologue: stage K-slab 0 into buffer 0  (6 x 16B per thread)
#pragma unroll
  for (int i = 0; i < 2; ++i) {
    const int c = tid + i * 256;              // A: 512 chunks of 8 halfs
    stage16(A + (rowBlock + (c >> 2)) * K + (c & 3) * 8,
            &As[0][c >> 2][(c & 3) * 8]);
  }
#pragma unroll
  for (int i = 0; i < 4; ++i) {
    const int c = tid + i * 256;              // B: 1024 chunks
    stage16(Bt + (colBlock + (c >> 2)) * K + (c & 3) * 8,
            &Bs[0][c >> 2][(c & 3) * 8]);
  }

  for (int t = 0; t < nk; ++t) {
    const int buf = t & 1;
    if (t + 1 < nk) {
      const int k0 = (t + 1) * 32, nb = buf ^ 1;
#pragma unroll
      for (int i = 0; i < 2; ++i) {
        const int c = tid + i * 256;
        stage16(A + (rowBlock + (c >> 2)) * K + k0 + (c & 3) * 8,
                &As[nb][c >> 2][(c & 3) * 8]);
      }
#pragma unroll
      for (int i = 0; i < 4; ++i) {
        const int c = tid + i * 256;
        stage16(Bt + (colBlock + (c >> 2)) * K + k0 + (c & 3) * 8,
                &Bs[nb][c >> 2][(c & 3) * 8]);
      }
      wait_async<6>();   // everything except the 6 just issued has landed
    } else {
      wait_async<0>();
    }
    __syncthreads();

    v16h af[4];
#pragma unroll
    for (int im = 0; im < 4; ++im)
      af[im] = frag_a_lds(&As[buf][0][0], 40, waveM * 64 + im * 16, 0, lane);
#pragma unroll
    for (int jn = 0; jn < 4; ++jn) {
      const v16h bf = frag_b_lds(&Bs[buf][0][0], 40, waveN * 64 + jn * 16, 0, lane);
#pragma unroll
      for (int im = 0; im < 4; ++im)
        acc[im][jn] = wmma_f16(af[im], bf, acc[im][jn]);
    }
    __syncthreads();   // readers done before this buffer is overwritten
  }

  const int hs = lane >> 4, ln = lane & 15;
#pragma unroll
  for (int im = 0; im < 4; ++im) {
#pragma unroll
    for (int jn = 0; jn < 4; ++jn) {
      const int col = (int)colBlock + waveN * 64 + jn * 16 + ln;
      const float bv = bias[col];
#pragma unroll
      for (int r = 0; r < 8; ++r) {
        const size_t rowg = rowBlock + waveM * 64 + im * 16 + r + 8 * hs;
        C[rowg * N + col] = acc[im][jn][r] + bv;
      }
    }
  }
}

// ---------------- Flash attention ----------------
// grid: (LQ/64, B*NH); block 128 = 4 waves, each wave owns 16 q rows.
// Qh/Kh: [B][H][L][HD] f16 (q pre-scaled by HD^-0.5); Vh: [B][H][HD][LKV] f16.
// K/V tiles double-buffered via async-to-LDS. Ah: (B*LQ, DIMM) f16.
__global__ __launch_bounds__(128) void flash_attn_kernel(
    const _Float16* __restrict__ Qh, const _Float16* __restrict__ Kh,
    const _Float16* __restrict__ Vh, _Float16* __restrict__ Ah) {
  __shared__ _Float16 Qs[64][72];
  __shared__ _Float16 Ks[2][64][72];
  __shared__ _Float16 Vt[2][64][72];   // [hd][kv]
  __shared__ _Float16 Ps[4][16][72];   // per-wave P scratch (C-layout -> A-layout)

  const int tid  = threadIdx.x;
  const int lane = tid & 31;
  const int wave = tid >> 5;
  const int bh = blockIdx.y;           // b*NH + h
  const int b  = bh >> 4;
  const int h  = bh & 15;
  const int qBase = blockIdx.x * 64;

  const _Float16* qptr = Qh + ((size_t)bh * LQ + qBase) * HD;
  const _Float16* kptr = Kh + (size_t)bh * LKV * HD;
  const _Float16* vptr = Vh + (size_t)bh * HD * LKV;

  // stage Q tile (4 x 16B per thread)
  for (int c = tid; c < 512; c += 128) {
    const int r = c >> 3, c8 = (c & 7) * 8;
    stage16(qptr + r * HD + c8, &Qs[r][c8]);
  }
  // stage K/V tile 0 (8 x 16B per thread)
  for (int c = tid; c < 512; c += 128) {
    const int r = c >> 3, c8 = (c & 7) * 8;
    stage16(kptr + (size_t)r * HD + c8, &Ks[0][r][c8]);
    stage16(vptr + (size_t)r * LKV + c8, &Vt[0][r][c8]);
  }

  v8f o[4];
  float m_r[8], l_r[8];
#pragma unroll
  for (int jo = 0; jo < 4; ++jo) { v8f z = {}; o[jo] = z; }
#pragma unroll
  for (int r = 0; r < 8; ++r) { m_r[r] = -1e30f; l_r[r] = 0.f; }

  for (int t = 0; t < LKV / 64; ++t) {
    const int buf = t & 1;
    if (t + 1 < LKV / 64) {
      const int kv = (t + 1) * 64, nb = buf ^ 1;
      for (int c = tid; c < 512; c += 128) {
        const int r = c >> 3, c8 = (c & 7) * 8;
        stage16(kptr + (size_t)(kv + r) * HD + c8, &Ks[nb][r][c8]);
        stage16(vptr + (size_t)r * LKV + kv + c8, &Vt[nb][r][c8]);
      }
      wait_async<8>();
    } else {
      wait_async<0>();
    }
    __syncthreads();

    // S = Q * K^T  (16x64 per wave)
    const v16h qa0 = frag_a_lds(&Qs[0][0], 72, wave * 16, 0, lane);
    const v16h qa1 = frag_a_lds(&Qs[0][0], 72, wave * 16, 32, lane);
    v8f s[4];
#pragma unroll
    for (int jn = 0; jn < 4; ++jn) {
      v8f z = {};
      z = wmma_f16(qa0, frag_b_lds(&Ks[buf][0][0], 72, jn * 16, 0, lane), z);
      z = wmma_f16(qa1, frag_b_lds(&Ks[buf][0][0], 72, jn * 16, 32, lane), z);
      s[jn] = z;
    }

    {
      const int hs = lane >> 4, ln = lane & 15;
#pragma unroll
      for (int r = 0; r < 8; ++r) {
        float v = fmaxf(fmaxf(s[0][r], s[1][r]), fmaxf(s[2][r], s[3][r]));
#pragma unroll
        for (int off = 1; off < 16; off <<= 1)
          v = fmaxf(v, __shfl_xor(v, off, 32));
        const float mnew  = fmaxf(m_r[r], v);
        const float alpha = __expf(m_r[r] - mnew);
        float rsum = 0.f;
#pragma unroll
        for (int jn = 0; jn < 4; ++jn) {
          const float pv = __expf(s[jn][r] - mnew);
          rsum += pv;
          Ps[wave][r + 8 * hs][jn * 16 + ln] = (_Float16)pv;
        }
#pragma unroll
        for (int off = 1; off < 16; off <<= 1)
          rsum += __shfl_xor(rsum, off, 32);
        l_r[r] = l_r[r] * alpha + rsum;
        m_r[r] = mnew;
#pragma unroll
        for (int jo = 0; jo < 4; ++jo) o[jo][r] = o[jo][r] * alpha;
      }
    }

    // wave-local LDS writes above feed the fragment reads below
    asm volatile("s_wait_dscnt 0" ::: "memory");

    // O += P * V
#pragma unroll
    for (int kc = 0; kc < 64; kc += 32) {
      const v16h pa = frag_a_lds(&Ps[wave][0][0], 72, 0, kc, lane);
#pragma unroll
      for (int jo = 0; jo < 4; ++jo)
        o[jo] = wmma_f16(pa, frag_b_lds(&Vt[buf][0][0], 72, jo * 16, kc, lane), o[jo]);
    }
    __syncthreads();   // readers done before buffer reuse
  }

  const int hs = lane >> 4, ln = lane & 15;
#pragma unroll
  for (int r = 0; r < 8; ++r) {
    const float inv = 1.f / l_r[r];
    const int rowg = qBase + wave * 16 + r + 8 * hs;
#pragma unroll
    for (int jo = 0; jo < 4; ++jo) {
      const int col = jo * 16 + ln;
      Ah[((size_t)(b * LQ) + rowg) * DIMM + h * HD + col] = (_Float16)(o[jo][r] * inv);
    }
  }
}

// ---------------- launcher ----------------

extern "C" void kernel_launch(void* const* d_in, const int* in_sizes, int n_in,
                              void* d_out, int out_size, void* d_ws, size_t ws_size,
                              hipStream_t stream) {
  (void)in_sizes; (void)n_in; (void)out_size; (void)ws_size;
  const float* x     = (const float*)d_in[0];
  const float* y     = (const float*)d_in[1];
  const float* x_cos = (const float*)d_in[2];
  const float* x_sin = (const float*)d_in[3];
  const float* y_cos = (const float*)d_in[4];
  const float* y_sin = (const float*)d_in[5];
  const float* Wq    = (const float*)d_in[6];
  const float* bq    = (const float*)d_in[7];
  const float* Wk    = (const float*)d_in[8];
  const float* bk    = (const float*)d_in[9];
  const float* Wv    = (const float*)d_in[10];
  const float* bv    = (const float*)d_in[11];
  const float* Wo    = (const float*)d_in[12];
  const float* bo    = (const float*)d_in[13];
  const float* gq    = (const float*)d_in[14];
  const float* gk    = (const float*)d_in[15];
  float* out = (float*)d_out;

  char* p = (char*)d_ws;
  auto alloc = [&](size_t bytes) -> char* {
    char* r = p;
    p += (bytes + 255) & ~(size_t)255;
    return r;
  };
  _Float16* Xh  = (_Float16*)alloc((size_t)BB * LQ * DIMM * 2);
  _Float16* Yh  = (_Float16*)alloc((size_t)BB * LKV * KVD * 2);
  _Float16* Wqt = (_Float16*)alloc((size_t)DIMM * DIMM * 2);
  _Float16* Wkt = (_Float16*)alloc((size_t)DIMM * KVD * 2);
  _Float16* Wvt = (_Float16*)alloc((size_t)DIMM * KVD * 2);
  _Float16* Wot = (_Float16*)alloc((size_t)DIMM * DIMM * 2);
  float*    Qf  = (float*)alloc((size_t)BB * LQ * DIMM * 4);
  float*    Kf  = (float*)alloc((size_t)BB * LKV * DIMM * 4);
  float*    Vf  = (float*)alloc((size_t)BB * LKV * DIMM * 4);
  _Float16* Qh  = (_Float16*)alloc((size_t)BB * LQ * DIMM * 2);
  _Float16* Kh  = (_Float16*)alloc((size_t)BB * LKV * DIMM * 2);
  _Float16* Vh  = (_Float16*)alloc((size_t)BB * LKV * DIMM * 2);
  _Float16* Ah  = (_Float16*)alloc((size_t)BB * LQ * DIMM * 2);

  // f32 -> f16 conversions (weights also transposed to N x K)
  cvt_f32_to_f16_kernel<<<(BB * LQ * DIMM) / 256, 256, 0, stream>>>(x, Xh, BB * LQ * DIMM);
  cvt_f32_to_f16_kernel<<<(BB * LKV * KVD) / 256, 256, 0, stream>>>(y, Yh, BB * LKV * KVD);
  cvt_transpose_kernel<<<(DIMM * DIMM) / 256, 256, 0, stream>>>(Wq, Wqt, DIMM, DIMM);
  cvt_transpose_kernel<<<(KVD * DIMM) / 256, 256, 0, stream>>>(Wk, Wkt, KVD, DIMM);
  cvt_transpose_kernel<<<(KVD * DIMM) / 256, 256, 0, stream>>>(Wv, Wvt, KVD, DIMM);
  cvt_transpose_kernel<<<(DIMM * DIMM) / 256, 256, 0, stream>>>(Wo, Wot, DIMM, DIMM);

  // QKV projections (WMMA, async double-buffered)
  gemm_f16_wmma_kernel<<<dim3(DIMM / 256, (BB * LQ) / 128), 256, 0, stream>>>(
      Xh, Wqt, bq, Qf, DIMM, DIMM);
  gemm_f16_wmma_kernel<<<dim3(DIMM / 256, (BB * LKV) / 128), 256, 0, stream>>>(
      Yh, Wkt, bk, Kf, DIMM, KVD);
  gemm_f16_wmma_kernel<<<dim3(DIMM / 256, (BB * LKV) / 128), 256, 0, stream>>>(
      Yh, Wvt, bv, Vf, DIMM, KVD);

  // RMSNorm + RoPE (softmax scale folded into q); V remapped to [B][H][HD][LKV]
  rmsnorm_rope_kernel<<<BB * LQ, 256, 0, stream>>>(Qf, gq, x_cos, x_sin, Qh, LQ, 0.125f);
  rmsnorm_rope_kernel<<<BB * LKV, 256, 0, stream>>>(Kf, gk, y_cos, y_sin, Kh, LKV, 1.0f);
  remap_v_kernel<<<(BB * LKV * DIMM) / 256, 256, 0, stream>>>(Vf, Vh);

  // Flash attention (WMMA for QK^T and PV, async double-buffered K/V)
  flash_attn_kernel<<<dim3(LQ / 64, BB * NH), 128, 0, stream>>>(Qh, Kh, Vh, Ah);

  // Output projection (WMMA), writes f32 result + bias to d_out
  gemm_f16_wmma_kernel<<<dim3(DIMM / 256, (BB * LQ) / 128), 256, 0, stream>>>(
      Ah, Wot, bo, out, DIMM, DIMM);
}
